// Attention_37160057044992
// MI455X (gfx1250) — matrix operand order repaired and checked by reading, not compiled, not run on hardware
//
#include <hip/hip_runtime.h>

// ---------------------------------------------------------------------------
// Types for CDNA5 WMMA bf16 path
// ---------------------------------------------------------------------------
typedef __attribute__((ext_vector_type(16))) __bf16 v16bf;
typedef __attribute__((ext_vector_type(8)))  __bf16 v8bf;
typedef __attribute__((ext_vector_type(8)))  float  v8f;
typedef __attribute__((ext_vector_type(4)))  float  v4f;

union ABfrag { v16bf v; v8bf h[2]; };

#define D_MODEL 4096
#define HQ      32
#define HKV     8
#define HD      128
#define BB      2
#define TT      2048

// ---------------------------------------------------------------------------
// GEMM: C[M x N] = A[M x K] (f32, row-major) @ B[K x N] (f32, row-major)
// f32 operands are down-converted to bf16 while staging into LDS, then fed to
// v_wmma_f32_16x16x32_bf16. Block tile 128x128, 8 waves, wave tile 16x128.
// blockIdx.z selects a head (B/C strided).
// ---------------------------------------------------------------------------
template <bool OUT_BF16>
__global__ __launch_bounds__(256)
void gemm_bf16_wmma(const float* __restrict__ A, long lda,
                    const float* __restrict__ Bsrc, long ldb, long strideBz,
                    void* __restrict__ Cdst, long ldc, long strideCz,
                    int K)
{
    const int m0   = blockIdx.x * 128;
    const int n0   = blockIdx.y * 128;
    const float* Bh = Bsrc + (long)blockIdx.z * strideBz;

    const int tid  = threadIdx.x;
    const int wave = tid >> 5;
    const int lane = tid & 31;
    const int Mloc = lane & 15;
    const int hi   = lane >> 4;            // 0: lanes 0-15, 1: lanes 16-31
    const int kbA  = hi ? 8 : 0;           // 16-bit A layout K-base
    const int kbB  = hi ? 8 : 0;           // 16-bit B layout K-base

    __shared__ alignas(16) __bf16 lA[128][40];   // [m][k], padded
    __shared__ alignas(16) __bf16 lB[128][40];   // [n][k] (transposed), padded

    v8f acc[8];
#pragma unroll
    for (int j = 0; j < 8; ++j)
#pragma unroll
        for (int i = 0; i < 8; ++i) acc[j][i] = 0.0f;

    const int arow  = tid >> 1;
    const int ahalf = (tid & 1) * 16;
    const int bk    = tid & 31;
    const int bn    = (tid >> 5) * 16;

    for (int k0 = 0; k0 < K; k0 += 32) {
        // ---- stage A tile (128 x 32), convert f32 -> bf16 ----
        {
            const v4f* src = (const v4f*)(A + (long)(m0 + arow) * lda + k0 + ahalf);
            v4f f0 = src[0], f1 = src[1], f2 = src[2], f3 = src[3];
            float ta[16];
            *(v4f*)&ta[0] = f0; *(v4f*)&ta[4] = f1;
            *(v4f*)&ta[8] = f2; *(v4f*)&ta[12] = f3;
            v8bf o0, o1;
#pragma unroll
            for (int i = 0; i < 8; ++i) { o0[i] = (__bf16)ta[i]; o1[i] = (__bf16)ta[8 + i]; }
            *(v8bf*)&lA[arow][ahalf]     = o0;
            *(v8bf*)&lA[arow][ahalf + 8] = o1;
        }
        // ---- stage B tile (32 x 128) transposed into [n][k] ----
        {
            const v4f* src = (const v4f*)(Bh + (long)(k0 + bk) * ldb + n0 + bn);
            v4f f0 = src[0], f1 = src[1], f2 = src[2], f3 = src[3];
            float tb[16];
            *(v4f*)&tb[0] = f0; *(v4f*)&tb[4] = f1;
            *(v4f*)&tb[8] = f2; *(v4f*)&tb[12] = f3;
#pragma unroll
            for (int i = 0; i < 16; ++i) lB[bn + i][bk] = (__bf16)tb[i];
        }
        if (k0 + 32 < K) {
            __builtin_prefetch(A + (long)(m0 + arow) * lda + k0 + 32 + ahalf, 0, 1);
            __builtin_prefetch(Bh + (long)(k0 + 32 + bk) * ldb + n0 + bn, 0, 1);
        }
        __syncthreads();

        // ---- wave tile: 16 rows x 128 cols, one A frag + 8 B frags ----
        ABfrag a;
        a.h[0] = *(const v8bf*)&lA[wave * 16 + Mloc][kbA];
        a.h[1] = *(const v8bf*)&lA[wave * 16 + Mloc][kbA + 16];
#pragma unroll
        for (int j = 0; j < 8; ++j) {
            ABfrag b;
            b.h[0] = *(const v8bf*)&lB[j * 16 + Mloc][kbB];
            b.h[1] = *(const v8bf*)&lB[j * 16 + Mloc][kbB + 16];
            acc[j] = __builtin_amdgcn_wmma_f32_16x16x32_bf16(
                false, a.v, false, b.v, (short)0, acc[j], false, false);
        }
        __syncthreads();
    }

    // ---- epilogue: C layout lanes 0-15 -> M=r, lanes 16-31 -> M=8+r ----
    const int mrow = m0 + wave * 16 + hi * 8;
    const int ncol = n0 + Mloc;
    if (OUT_BF16) {
        __bf16* C = (__bf16*)Cdst + (long)blockIdx.z * strideCz;
#pragma unroll
        for (int r = 0; r < 8; ++r)
#pragma unroll
            for (int j = 0; j < 8; ++j)
                C[(long)(mrow + r) * ldc + ncol + 16 * j] = (__bf16)acc[j][r];
    } else {
        float* C = (float*)Cdst + (long)blockIdx.z * strideCz;
#pragma unroll
        for (int r = 0; r < 8; ++r)
#pragma unroll
            for (int j = 0; j < 8; ++j)
                C[(long)(mrow + r) * ldc + ncol + 16 * j] = acc[j][r];
    }
}

// ---------------------------------------------------------------------------
// RoPE in place on bf16 buffer laid out [B, T, nheads, HD]
// ---------------------------------------------------------------------------
__global__ __launch_bounds__(256)
void rope_kernel(__bf16* __restrict__ buf, const int* __restrict__ positions,
                 int nheads, int total)
{
    int i = blockIdx.x * blockDim.x + threadIdx.x;
    if (i >= total) return;
    int h = i & 63;
    int rest = i >> 6;
    int n = rest % nheads;
    int t = (rest / nheads) % TT;
    int b = rest / (nheads * TT);
    long base = (((long)(b * TT + t)) * nheads + n) * HD;
    float first  = (float)buf[base + h];
    float second = (float)buf[base + h + 64];
    float frac = (2.0f * (float)h) * (1.0f / 128.0f);
    float ts = __expf(frac * 9.210340371976184f);   // theta^frac, ln(10000)
    float ang = (float)positions[b * TT + t] / ts;
    float s, c;
    __sincosf(ang, &s, &c);
    buf[base + h]      = (__bf16)(first * c - second * s);
    buf[base + h + 64] = (__bf16)(second * c + first * s);
}

// ---------------------------------------------------------------------------
// Flash-style causal GQA attention.
// grid.x = B*HQ (same kv head for whole block), grid.y = T/128.
// 8 waves/block, each wave owns a 16-row query tile; K/V tiles of 32 keys are
// cooperatively staged in LDS (V transposed). P goes through per-wave LDS to
// convert the WMMA C layout into the A layout for the P@V WMMAs.
// ---------------------------------------------------------------------------
__global__ __launch_bounds__(256)
void attn_kernel(const __bf16* __restrict__ q, const __bf16* __restrict__ k,
                 const __bf16* __restrict__ v, float* __restrict__ o)
{
    const int bh   = blockIdx.x;
    const int b    = bh / HQ;
    const int n    = bh % HQ;
    const int kvn  = n / (HQ / HKV);
    const int wave = threadIdx.x >> 5;
    const int lane = threadIdx.x & 31;
    const int t0   = (blockIdx.y * 8 + wave) * 16;
    const int Nloc = lane & 15;
    const int hi   = lane >> 4;
    const int kbA  = hi ? 8 : 0;
    const int kbB  = hi ? 8 : 0;
    const float scale = 0.08838834764831845f;    // 1/sqrt(128)

    __shared__ alignas(16) __bf16 lK[32][136];      // [s][h]
    __shared__ alignas(16) __bf16 lV[128][40];      // [h][s] (transposed)
    __shared__ alignas(16) __bf16 lP[8][16][32];    // per-wave P tile

    // Q fragments (16 x 128 split into 4 A-frags of 16x32), live in VGPRs.
    ABfrag aq[4];
    {
        const long qrow = ((long)(b * TT + t0 + Nloc)) * (HQ * HD) + (long)n * HD;
#pragma unroll
        for (int kk = 0; kk < 4; ++kk) {
            aq[kk].h[0] = *(const v8bf*)(q + qrow + kk * 32 + kbA);
            aq[kk].h[1] = *(const v8bf*)(q + qrow + kk * 32 + kbA + 16);
        }
    }

    v8f acco[8];
    float rm[8], ls[8];
#pragma unroll
    for (int j = 0; j < 8; ++j) {
#pragma unroll
        for (int i = 0; i < 8; ++i) acco[j][i] = 0.0f;
        rm[j] = -3.0e38f;
        ls[j] = 0.0f;
    }

    const int  nSblk  = (blockIdx.y + 1) * 4;           // uniform over block
    const long kvbase = (long)b * TT * (HKV * HD) + (long)kvn * HD;

    for (int sb = 0; sb < nSblk; ++sb) {
        const int s0 = sb * 32;
        __syncthreads();
        // ---- stage K tile (32 x 128) ----
        {
            int sr = threadIdx.x >> 3, ch = (threadIdx.x & 7) * 16;
            const __bf16* src = k + kvbase + (long)(s0 + sr) * (HKV * HD) + ch;
            *(v8bf*)&lK[sr][ch]     = *(const v8bf*)(src);
            *(v8bf*)&lK[sr][ch + 8] = *(const v8bf*)(src + 8);
        }
        // ---- stage V tile transposed: lV[h][s] ----
        {
            int sr = threadIdx.x & 31, hch = (threadIdx.x >> 5) * 16;
            const __bf16* src = v + kvbase + (long)(s0 + sr) * (HKV * HD) + hch;
            v8bf a0 = *(const v8bf*)(src);
            v8bf a1 = *(const v8bf*)(src + 8);
#pragma unroll
            for (int i = 0; i < 8; ++i) lV[hch + i][sr] = a0[i];
#pragma unroll
            for (int i = 0; i < 8; ++i) lV[hch + 8 + i][sr] = a1[i];
        }
        __syncthreads();

        // ---- scores: two 16-key subtiles ----
        v8f sc[2];
#pragma unroll
        for (int sub = 0; sub < 2; ++sub) {
            v8f s;
#pragma unroll
            for (int i = 0; i < 8; ++i) s[i] = 0.0f;
#pragma unroll
            for (int kk = 0; kk < 4; ++kk) {
                ABfrag bk;
                bk.h[0] = *(const v8bf*)&lK[sub * 16 + Nloc][kk * 32 + kbB];
                bk.h[1] = *(const v8bf*)&lK[sub * 16 + Nloc][kk * 32 + kbB + 16];
                s = __builtin_amdgcn_wmma_f32_16x16x32_bf16(
                    false, aq[kk].v, false, bk.v, (short)0, s, false, false);
            }
            sc[sub] = s;
        }

        // ---- causal mask + online softmax + P -> LDS (bf16, A-layout src) ----
#pragma unroll
        for (int r = 0; r < 8; ++r) {
            int m    = r + hi * 8;
            int tpos = t0 + m;
            int sp0  = s0 + Nloc;
            int sp1  = s0 + 16 + Nloc;
            float v0 = (sp0 <= tpos) ? sc[0][r] * scale : -3.0e38f;
            float v1 = (sp1 <= tpos) ? sc[1][r] * scale : -3.0e38f;
            float mx = fmaxf(v0, v1);
#pragma unroll
            for (int off = 8; off >= 1; off >>= 1)
                mx = fmaxf(mx, __shfl_xor(mx, off, 32));
            float mnew = fmaxf(rm[r], mx);
            float corr = __expf(rm[r] - mnew);
            float p0 = __expf(v0 - mnew);
            float p1 = __expf(v1 - mnew);
            float rs = p0 + p1;
#pragma unroll
            for (int off = 8; off >= 1; off >>= 1)
                rs += __shfl_xor(rs, off, 32);
            ls[r] = ls[r] * corr + rs;
            rm[r] = mnew;
#pragma unroll
            for (int j = 0; j < 8; ++j) acco[j][r] *= corr;
            lP[wave][m][Nloc]      = (__bf16)p0;
            lP[wave][m][16 + Nloc] = (__bf16)p1;
        }

        // ---- P @ V over the 32-key block (per-wave LDS, DS in-order) ----
        ABfrag ap;
        ap.h[0] = *(const v8bf*)&lP[wave][Nloc][kbA];
        ap.h[1] = *(const v8bf*)&lP[wave][Nloc][kbA + 16];
#pragma unroll
        for (int j = 0; j < 8; ++j) {
            ABfrag bv;
            bv.h[0] = *(const v8bf*)&lV[j * 16 + Nloc][kbB];
            bv.h[1] = *(const v8bf*)&lV[j * 16 + Nloc][kbB + 16];
            acco[j] = __builtin_amdgcn_wmma_f32_16x16x32_bf16(
                false, ap.v, false, bv.v, (short)0, acco[j], false, false);
        }
    }

    // ---- epilogue: o[b,t,n,h] in f32, row-major over (n*128+h) ----
#pragma unroll
    for (int r = 0; r < 8; ++r) {
        int m = r + hi * 8;
        long orow = (((long)(b * TT + t0 + m)) * HQ + n) * HD;
        float inv = 1.0f / ls[r];
#pragma unroll
        for (int j = 0; j < 8; ++j)
            o[orow + j * 16 + Nloc] = acco[j][r] * inv;
    }
}

// ---------------------------------------------------------------------------
// Launcher
// ---------------------------------------------------------------------------
extern "C" void kernel_launch(void* const* d_in, const int* in_sizes, int n_in,
                              void* d_out, int out_size, void* d_ws, size_t ws_size,
                              hipStream_t stream) {
    (void)in_sizes; (void)n_in; (void)out_size; (void)ws_size;
    const float* x  = (const float*)d_in[0];
    const float* wq = (const float*)d_in[1];
    const float* wk = (const float*)d_in[2];
    const float* wv = (const float*)d_in[3];
    const float* wo = (const float*)d_in[4];
    const int*   pos = (const int*)d_in[5];
    float* out = (float*)d_out;

    char* ws = (char*)d_ws;
    __bf16* qb = (__bf16*)ws;                                   // [B*T, 32*128] bf16
    __bf16* kb = (__bf16*)(ws + 33554432);                      // [B*T,  8*128] bf16
    __bf16* vb = (__bf16*)(ws + 33554432 + 8388608);            // [B*T,  8*128] bf16
    float*  ob = (float*)(ws + 33554432 + 2 * 8388608);         // [B*T, 32*128] f32

    const dim3 blk(256);
    const int M = BB * TT;                 // 4096 tokens
    const long headW = (long)D_MODEL * HD; // weight head stride

    // QKV projections (f32 in, bf16 out, token-major [b,t,n,h])
    gemm_bf16_wmma<true><<<dim3(M / 128, 1, HQ),  blk, 0, stream>>>(
        x, D_MODEL, wq, HD, headW, qb, HQ * HD,  HD, D_MODEL);
    gemm_bf16_wmma<true><<<dim3(M / 128, 1, HKV), blk, 0, stream>>>(
        x, D_MODEL, wk, HD, headW, kb, HKV * HD, HD, D_MODEL);
    gemm_bf16_wmma<true><<<dim3(M / 128, 1, HKV), blk, 0, stream>>>(
        x, D_MODEL, wv, HD, headW, vb, HKV * HD, HD, D_MODEL);

    // RoPE on q and k (in place)
    int totq = BB * TT * HQ * 64;
    int totk = BB * TT * HKV * 64;
    rope_kernel<<<(totq + 255) / 256, blk, 0, stream>>>(qb, pos, HQ, totq);
    rope_kernel<<<(totk + 255) / 256, blk, 0, stream>>>(kb, pos, HKV, totk);

    // Causal GQA attention -> ob (f32)
    attn_kernel<<<dim3(BB * HQ, TT / 128), blk, 0, stream>>>(qb, kb, vb, ob);

    // Output projection: out[t,d] = ob[t,:] @ wo_flat[:, d]
    gemm_bf16_wmma<false><<<dim3(M / 128, D_MODEL / 128, 1), blk, 0, stream>>>(
        ob, D_MODEL, wo, D_MODEL, 0, out, D_MODEL, 0, D_MODEL);
}
